// Decoder_59133109732072
// MI455X (gfx1250) — matrix operand order repaired and checked by reading, not compiled, not run on hardware
//
#include <hip/hip_runtime.h>
#include <hip/hip_bf16.h>
#include <math.h>

// ---------------------------------------------------------------------------
// Types / constants
// ---------------------------------------------------------------------------
typedef __bf16 bf16_t;
typedef __attribute__((ext_vector_type(16))) __bf16 v16bf;
typedef __attribute__((ext_vector_type(8)))  float  v8f;

#define NB   32      // persistent decoder blocks (256 waves x 2 N-tiles each)
#define BT   256     // threads per block (8 wave32 per block)
#define TT   800     // time steps
#define BB   32      // batch
#define HH   1024    // RNN hidden
#define G4   4096    // 4*H
#define ENC  544     // duration feature dim
#define XC   800     // concat(prenet 256, dur 544)
#define PRE  256
#define KP   96      // padded prenet input K (80 -> 96)
#define KOUT 1568    // RNN + ENC for output projection
#define NMEL 80

// ---------------------------------------------------------------------------
// Device-scope barrier for the persistent decoder kernel
// ---------------------------------------------------------------------------
__device__ unsigned g_bar_count = 0;
__device__ unsigned g_bar_gen   = 0;

__device__ __forceinline__ void grid_sync() {
  __syncthreads();
  __threadfence();
  if (threadIdx.x == 0) {
    volatile unsigned* vgen = (volatile unsigned*)&g_bar_gen;
    unsigned gen  = *vgen;
    unsigned prev = atomicAdd(&g_bar_count, 1u);
    if (prev == NB - 1u) {
      g_bar_count = 0u;
      __threadfence();
      atomicAdd(&g_bar_gen, 1u);
    } else {
      while (*vgen == gen) { __builtin_amdgcn_s_sleep(1); }
    }
  }
  __threadfence();
  __syncthreads();
}

// ---------------------------------------------------------------------------
// WMMA helpers (wave32, 16x16x32 bf16 -> f32).  Layout per CDNA5 ISA 7.12.2:
//  A 16x32: lane m=lane&15, half=lane>>4;
//     elem i<8 -> K = half*8 + i ; elem i>=8 -> K = 16 + half*8 + (i-8)
//  B 32x16: lane n=lane&15, half=lane>>4, same K mapping as A
//  C/D f32: row = half*8 + r, col = lane&15
// ---------------------------------------------------------------------------
__device__ __forceinline__ v8f wmma_bf16(v16bf a, v16bf b, v8f c) {
  return __builtin_amdgcn_wmma_f32_16x16x32_bf16(false, a, false, b,
                                                 (short)0, c, false, false);
}

__device__ __forceinline__ v16bf load_a(const bf16_t* __restrict__ A, int lda,
                                        int m, int half, int k0) {
  const bf16_t* p = A + (size_t)m * lda + k0 + half * 8;
  v16bf v;
#pragma unroll
  for (int i = 0; i < 8; ++i) v[i] = p[i];
#pragma unroll
  for (int i = 0; i < 8; ++i) v[i + 8] = p[16 + i];
  return v;
}

// W is [N, K] row-major (torch Linear weight); B[k][n] = W[n*ldk + k]
__device__ __forceinline__ v16bf load_b(const bf16_t* __restrict__ W, int ldk,
                                        int n, int half, int k0) {
  const bf16_t* p = W + (size_t)n * ldk + k0 + half * 8;
  v16bf v;
#pragma unroll
  for (int i = 0; i < 8; ++i) v[i] = p[i];
#pragma unroll
  for (int i = 0; i < 8; ++i) v[i + 8] = p[16 + i];
  return v;
}

// Pipelined region, dual N-tile: on entry (a,b0,b1) hold tile 0 of this
// region; on exit they hold tile nkt-1 (not yet multiplied).
__device__ __forceinline__ void gemm2_region(
    v8f& acc0, v8f& acc1, v16bf& a, v16bf& b0, v16bf& b1,
    const bf16_t* __restrict__ Abase, int lda, int aRow,
    const bf16_t* __restrict__ Wbase, int ldk, int c0, int c1, int half,
    int nkt) {
  for (int kt = 1; kt < nkt; ++kt) {
    v16bf an  = load_a(Abase, lda, aRow, half, kt * 32);
    v16bf b0n = load_b(Wbase, ldk, c0, half, kt * 32);
    v16bf b1n = load_b(Wbase, ldk, c1, half, kt * 32);
    acc0 = wmma_bf16(a, b0, acc0);
    acc1 = wmma_bf16(a, b1, acc1);
    a = an; b0 = b0n; b1 = b1n;
  }
}

// Pipelined region, single N-tile (kofsB lets B use a shifted K origin).
__device__ __forceinline__ void gemm1_region(
    v8f& acc, v16bf& a, v16bf& b,
    const bf16_t* __restrict__ Abase, int lda, int aRow,
    const bf16_t* __restrict__ Wbase, int ldk, int c0, int kofsB, int half,
    int nkt) {
  for (int kt = 1; kt < nkt; ++kt) {
    v16bf an = load_a(Abase, lda, aRow, half, kt * 32);
    v16bf bn = load_b(Wbase, ldk, c0, half, kofsB + kt * 32);
    acc = wmma_bf16(a, b, acc);
    a = an; b = bn;
  }
}

__device__ __forceinline__ float sigm(float x) {
  return 1.0f / (1.0f + __expf(-x));
}

// ---------------------------------------------------------------------------
// f32 -> bf16 conversion / layout-prep kernels
// ---------------------------------------------------------------------------
__global__ void k_cvt(const float* __restrict__ s, bf16_t* __restrict__ d, int n) {
  for (int i = blockIdx.x * blockDim.x + threadIdx.x; i < n;
       i += gridDim.x * blockDim.x)
    d[i] = (bf16_t)s[i];
}

// W1 [256,80] -> padded [256,96]
__global__ void k_cvt_w1(const float* __restrict__ W1, bf16_t* __restrict__ d) {
  const int n = PRE * KP;
  for (int i = blockIdx.x * blockDim.x + threadIdx.x; i < n;
       i += gridDim.x * blockDim.x) {
    int o = i / KP, k = i % KP;
    d[i] = (bf16_t)((k < NMEL) ? W1[o * NMEL + k] : 0.0f);
  }
}

// decoder_inputs [B,80,T] -> decp [T,B,96] with go-frame shift and zero pad
__global__ void k_prep_dec(const float* __restrict__ dec, bf16_t* __restrict__ dp) {
  const int n = TT * BB * KP;
  for (int i = blockIdx.x * blockDim.x + threadIdx.x; i < n;
       i += gridDim.x * blockDim.x) {
    int k = i % KP;
    int b = (i / KP) & (BB - 1);
    int t = i / (KP * BB);
    float v = 0.0f;
    if (k < NMEL && t > 0) v = dec[((size_t)b * NMEL + k) * TT + (t - 1)];
    dp[i] = (bf16_t)v;
  }
}

// duration_outputs [B,T,544] -> xcat[T,B,800] columns 256..799
__global__ void k_prep_dur(const float* __restrict__ du, bf16_t* __restrict__ xcat) {
  const int n = TT * BB * ENC;
  for (int i = blockIdx.x * blockDim.x + threadIdx.x; i < n;
       i += gridDim.x * blockDim.x) {
    int j = i % ENC;
    int b = (i / ENC) & (BB - 1);
    int t = i / (ENC * BB);
    xcat[((size_t)t * BB + b) * XC + PRE + j] =
        (bf16_t)du[((size_t)b * TT + t) * ENC + j];
  }
}

// ---------------------------------------------------------------------------
// Prenet stage 1: a1 = relu(decp @ W1b^T) * pm1   (25600 x 96 -> 25600 x 256)
// ---------------------------------------------------------------------------
__global__ __launch_bounds__(BT) void k_prenet1(const bf16_t* __restrict__ decp,
                                                const bf16_t* __restrict__ W1b,
                                                const float* __restrict__ pm1,
                                                bf16_t* __restrict__ a1) {
  const int lane = threadIdx.x & 31, half = lane >> 4, l16 = lane & 15;
  const int gw = (blockIdx.x * blockDim.x + threadIdx.x) >> 5;
  const int nw = (gridDim.x * blockDim.x) >> 5;
  const int ntiles = (TT * BB / 16) * (PRE / 16);
  for (int tile = gw; tile < ntiles; tile += nw) {
    int mt = tile >> 4, nt = tile & 15;
    v8f acc;
#pragma unroll
    for (int r = 0; r < 8; ++r) acc[r] = 0.0f;
    v16bf a = load_a(decp, KP, mt * 16 + l16, half, 0);
    v16bf b = load_b(W1b, KP, nt * 16 + l16, half, 0);
    gemm1_region(acc, a, b, decp, KP, mt * 16 + l16, W1b, KP, nt * 16 + l16, 0,
                 half, KP / 32);
    acc = wmma_bf16(a, b, acc);
#pragma unroll
    for (int r = 0; r < 8; ++r) {
      int row = mt * 16 + half * 8 + r, c = nt * 16 + l16;
      float v = fmaxf(acc[r], 0.0f) * pm1[(size_t)row * PRE + c];
      a1[(size_t)row * PRE + c] = (bf16_t)v;
    }
  }
}

// ---------------------------------------------------------------------------
// Prenet stage 2: xcat[:, :256] = relu(a1 @ W2b^T) * pm2
// ---------------------------------------------------------------------------
__global__ __launch_bounds__(BT) void k_prenet2(const bf16_t* __restrict__ a1,
                                                const bf16_t* __restrict__ W2b,
                                                const float* __restrict__ pm2,
                                                bf16_t* __restrict__ xcat) {
  const int lane = threadIdx.x & 31, half = lane >> 4, l16 = lane & 15;
  const int gw = (blockIdx.x * blockDim.x + threadIdx.x) >> 5;
  const int nw = (gridDim.x * blockDim.x) >> 5;
  const int ntiles = (TT * BB / 16) * (PRE / 16);
  for (int tile = gw; tile < ntiles; tile += nw) {
    int mt = tile >> 4, nt = tile & 15;
    v8f acc;
#pragma unroll
    for (int r = 0; r < 8; ++r) acc[r] = 0.0f;
    v16bf a = load_a(a1, PRE, mt * 16 + l16, half, 0);
    v16bf b = load_b(W2b, PRE, nt * 16 + l16, half, 0);
    gemm1_region(acc, a, b, a1, PRE, mt * 16 + l16, W2b, PRE, nt * 16 + l16, 0,
                 half, PRE / 32);
    acc = wmma_bf16(a, b, acc);
#pragma unroll
    for (int r = 0; r < 8; ++r) {
      int row = mt * 16 + half * 8 + r, c = nt * 16 + l16;
      float v = fmaxf(acc[r], 0.0f) * pm2[(size_t)row * PRE + c];
      xcat[(size_t)row * XC + c] = (bf16_t)v;
    }
  }
}

// ---------------------------------------------------------------------------
// LayerNorm + LSTM pointwise update for one batch row (one block, 256 thr)
// ---------------------------------------------------------------------------
__device__ __forceinline__ void cell_update(int bIdx, int t,
                                            const float* __restrict__ gates,
                                            const float* __restrict__ g,
                                            const float* __restrict__ be,
                                            float* __restrict__ c,
                                            bf16_t* __restrict__ h,
                                            const float* __restrict__ dm,
                                            bf16_t* __restrict__ hist,
                                            float* r1, float* r2) {
  const int tid = threadIdx.x;
  const float* gr = gates + (size_t)bIdx * G4;
  float s1 = 0.0f, s2 = 0.0f;
#pragma unroll
  for (int u = 0; u < 16; ++u) {
    float z = gr[tid * 16 + u];
    s1 += z;
    s2 += z * z;
  }
  r1[tid] = s1;
  r2[tid] = s2;
  __syncthreads();
  for (int s = BT / 2; s > 0; s >>= 1) {
    if (tid < s) { r1[tid] += r1[tid + s]; r2[tid] += r2[tid + s]; }
    __syncthreads();
  }
  float mean = r1[0] * (1.0f / (float)G4);
  float var  = fmaxf(r2[0] * (1.0f / (float)G4) - mean * mean, 0.0f);
  float rs   = rsqrtf(var + 1e-5f);
#pragma unroll
  for (int q = 0; q < HH / BT; ++q) {
    int j = tid + q * BT;
    float zi = (gr[j]          - mean) * rs * g[j]          + be[j];
    float zf = (gr[HH + j]     - mean) * rs * g[HH + j]     + be[HH + j];
    float zg = (gr[2 * HH + j] - mean) * rs * g[2 * HH + j] + be[2 * HH + j];
    float zo = (gr[3 * HH + j] - mean) * rs * g[3 * HH + j] + be[3 * HH + j];
    float cn = sigm(zf) * c[bIdx * HH + j] + sigm(zi) * tanhf(zg);
    float hv = sigm(zo) * tanhf(cn);
    hv *= dm[((size_t)t * BB + bIdx) * HH + j];
    c[bIdx * HH + j] = cn;
    bf16_t hb = (bf16_t)hv;
    h[bIdx * HH + j] = hb;
    if (hist) hist[((size_t)t * BB + bIdx) * HH + j] = hb;
  }
}

// ---------------------------------------------------------------------------
// Persistent sequential decoder.  256 waves; each wave owns TWO 16x16 output
// tiles sharing one A tile (halved A traffic, doubled WMMA density); 1-deep
// software pipeline keeps loads in flight under the WMMAs.
// ---------------------------------------------------------------------------
__global__ __launch_bounds__(BT) void k_decoder(
    const bf16_t* __restrict__ xcat,
    const bf16_t* __restrict__ Wih1b, const bf16_t* __restrict__ Whh1b,
    const bf16_t* __restrict__ Wih2b, const bf16_t* __restrict__ Whh2b,
    const float* __restrict__ b1, const float* __restrict__ g1,
    const float* __restrict__ be1, const float* __restrict__ b2,
    const float* __restrict__ g2, const float* __restrict__ be2,
    const float* __restrict__ dm1, const float* __restrict__ dm2,
    bf16_t* __restrict__ h1b, bf16_t* __restrict__ h2b,
    float* __restrict__ c1, float* __restrict__ c2,
    float* __restrict__ gates, bf16_t* __restrict__ h2hist) {
  __shared__ float r1[BT], r2[BT];
  const int tid  = threadIdx.x;
  const int lane = tid & 31, half = lane >> 4, l16 = lane & 15;
  const int wid  = blockIdx.x * (BT / 32) + (tid >> 5);  // 0..255
  const int mt   = wid >> 7;                             // 0..1
  const int ntp  = wid & 127;                            // 0..127
  const int c0   = ntp * 32 + l16;                       // first N column
  const int c1c  = c0 + 16;                              // second N column
  const int rowM = mt * 16 + half * 8;
  const int aRow = mt * 16 + l16;

  // zero recurrent state each launch (ws is poisoned, not re-zeroed)
  for (int i = blockIdx.x * BT + tid; i < BB * HH; i += NB * BT) {
    h1b[i] = (bf16_t)0.0f;
    h2b[i] = (bf16_t)0.0f;
    c1[i] = 0.0f;
    c2[i] = 0.0f;
  }
  grid_sync();

  for (int t = 0; t < TT; ++t) {
    // ---- gates1 = [x_t | dur_t] @ Wih1^T + h1 @ Whh1^T + b1 ----
    {
      const bf16_t* xr = xcat + (size_t)t * BB * XC;
      float bc0 = b1[c0], bc1 = b1[c1c];
      v8f acc0, acc1;
#pragma unroll
      for (int r = 0; r < 8; ++r) { acc0[r] = bc0; acc1[r] = bc1; }
      v16bf a  = load_a(xr, XC, aRow, half, 0);
      v16bf b0 = load_b(Wih1b, XC, c0, half, 0);
      v16bf b1v = load_b(Wih1b, XC, c1c, half, 0);
      gemm2_region(acc0, acc1, a, b0, b1v, xr, XC, aRow, Wih1b, XC, c0, c1c,
                   half, XC / 32);
      {  // handoff: load first h tile, consume last x tile
        v16bf an  = load_a(h1b, HH, aRow, half, 0);
        v16bf b0n = load_b(Whh1b, HH, c0, half, 0);
        v16bf b1n = load_b(Whh1b, HH, c1c, half, 0);
        acc0 = wmma_bf16(a, b0, acc0);
        acc1 = wmma_bf16(a, b1v, acc1);
        a = an; b0 = b0n; b1v = b1n;
      }
      gemm2_region(acc0, acc1, a, b0, b1v, h1b, HH, aRow, Whh1b, HH, c0, c1c,
                   half, HH / 32);
      acc0 = wmma_bf16(a, b0, acc0);
      acc1 = wmma_bf16(a, b1v, acc1);
#pragma unroll
      for (int r = 0; r < 8; ++r) {
        gates[(size_t)(rowM + r) * G4 + c0]  = acc0[r];
        gates[(size_t)(rowM + r) * G4 + c1c] = acc1[r];
      }
    }
    grid_sync();

    // ---- cell 1 (LN + LSTM pointwise), one block per batch row ----
    if (blockIdx.x < BB)
      cell_update(blockIdx.x, t, gates, g1, be1, c1, h1b, dm1, nullptr, r1, r2);
    grid_sync();

    // ---- gates2 = h1 @ Wih2^T + h2 @ Whh2^T + b2 ----
    {
      float bc0 = b2[c0], bc1 = b2[c1c];
      v8f acc0, acc1;
#pragma unroll
      for (int r = 0; r < 8; ++r) { acc0[r] = bc0; acc1[r] = bc1; }
      v16bf a  = load_a(h1b, HH, aRow, half, 0);
      v16bf b0 = load_b(Wih2b, HH, c0, half, 0);
      v16bf b1v = load_b(Wih2b, HH, c1c, half, 0);
      gemm2_region(acc0, acc1, a, b0, b1v, h1b, HH, aRow, Wih2b, HH, c0, c1c,
                   half, HH / 32);
      {  // handoff into the h2 @ Whh2 region
        v16bf an  = load_a(h2b, HH, aRow, half, 0);
        v16bf b0n = load_b(Whh2b, HH, c0, half, 0);
        v16bf b1n = load_b(Whh2b, HH, c1c, half, 0);
        acc0 = wmma_bf16(a, b0, acc0);
        acc1 = wmma_bf16(a, b1v, acc1);
        a = an; b0 = b0n; b1v = b1n;
      }
      gemm2_region(acc0, acc1, a, b0, b1v, h2b, HH, aRow, Whh2b, HH, c0, c1c,
                   half, HH / 32);
      acc0 = wmma_bf16(a, b0, acc0);
      acc1 = wmma_bf16(a, b1v, acc1);
#pragma unroll
      for (int r = 0; r < 8; ++r) {
        gates[(size_t)(rowM + r) * G4 + c0]  = acc0[r];
        gates[(size_t)(rowM + r) * G4 + c1c] = acc1[r];
      }
    }
    grid_sync();

    // ---- cell 2, also record h2 into the history for the final projection
    if (blockIdx.x < BB)
      cell_update(blockIdx.x, t, gates, g2, be2, c2, h2b, dm2, h2hist, r1, r2);
    grid_sync();
  }
}

// ---------------------------------------------------------------------------
// Final projection: out[b,mel,t] = [h2(t,b) | dur(t,b)] @ Wp^T + bp
// ---------------------------------------------------------------------------
__global__ __launch_bounds__(BT) void k_outproj(const bf16_t* __restrict__ h2hist,
                                                const bf16_t* __restrict__ xcat,
                                                const bf16_t* __restrict__ Wpb,
                                                const float* __restrict__ bp,
                                                float* __restrict__ out) {
  const int lane = threadIdx.x & 31, half = lane >> 4, l16 = lane & 15;
  const int gw = (blockIdx.x * blockDim.x + threadIdx.x) >> 5;
  const int nw = (gridDim.x * blockDim.x) >> 5;
  const int ntiles = (TT * BB / 16) * (NMEL / 16);
  for (int tile = gw; tile < ntiles; tile += nw) {
    int mt = tile / (NMEL / 16), nt = tile % (NMEL / 16);
    int aR = mt * 16 + l16, cN = nt * 16 + l16;
    float bc = bp[cN];
    v8f acc;
#pragma unroll
    for (int r = 0; r < 8; ++r) acc[r] = bc;
    v16bf a = load_a(h2hist, HH, aR, half, 0);
    v16bf b = load_b(Wpb, KOUT, cN, half, 0);
    gemm1_region(acc, a, b, h2hist, HH, aR, Wpb, KOUT, cN, 0, half, HH / 32);
    {  // handoff: start dur region, consume last h2 tile
      v16bf an = load_a(xcat + PRE, XC, aR, half, 0);
      v16bf bn = load_b(Wpb, KOUT, cN, half, HH);
      acc = wmma_bf16(a, b, acc);
      a = an; b = bn;
    }
    gemm1_region(acc, a, b, xcat + PRE, XC, aR, Wpb, KOUT, cN, HH, half,
                 ENC / 32);
    acc = wmma_bf16(a, b, acc);
#pragma unroll
    for (int r = 0; r < 8; ++r) {
      int row = mt * 16 + half * 8 + r;
      int t = row >> 5, b2 = row & (BB - 1);
      out[((size_t)b2 * NMEL + cN) * TT + t] = acc[r];
    }
  }
}

// ---------------------------------------------------------------------------
// Host launcher
// ---------------------------------------------------------------------------
extern "C" void kernel_launch(void* const* d_in, const int* in_sizes, int n_in,
                              void* d_out, int out_size, void* d_ws,
                              size_t ws_size, hipStream_t stream) {
  (void)in_sizes; (void)n_in; (void)out_size; (void)ws_size;
  const float* p_dur  = (const float*)d_in[0];   // [32,800,544]
  const float* p_dec  = (const float*)d_in[1];   // [32,80,800]
  // d_in[2] memory_lengths unused by the reference math
  const float* W1  = (const float*)d_in[3];
  const float* W2  = (const float*)d_in[4];
  const float* Wih1 = (const float*)d_in[5];
  const float* Whh1 = (const float*)d_in[6];
  const float* b1  = (const float*)d_in[7];
  const float* g1  = (const float*)d_in[8];
  const float* be1 = (const float*)d_in[9];
  const float* Wih2 = (const float*)d_in[10];
  const float* Whh2 = (const float*)d_in[11];
  const float* b2  = (const float*)d_in[12];
  const float* g2  = (const float*)d_in[13];
  const float* be2 = (const float*)d_in[14];
  const float* Wp  = (const float*)d_in[15];
  const float* bp  = (const float*)d_in[16];
  const float* pm1 = (const float*)d_in[17];
  const float* pm2 = (const float*)d_in[18];
  const float* dm1 = (const float*)d_in[19];
  const float* dm2 = (const float*)d_in[20];
  float* out = (float*)d_out;

  // workspace carve-up (all offsets 256B aligned)
  char* ws = (char*)d_ws;
  size_t off = 0;
  auto carve = [&](size_t bytes) -> char* {
    char* p = ws + off;
    off = (off + bytes + 255) & ~(size_t)255;
    return p;
  };
  bf16_t* Wih1b  = (bf16_t*)carve((size_t)G4 * XC * 2);
  bf16_t* Whh1b  = (bf16_t*)carve((size_t)G4 * HH * 2);
  bf16_t* Wih2b  = (bf16_t*)carve((size_t)G4 * HH * 2);
  bf16_t* Whh2b  = (bf16_t*)carve((size_t)G4 * HH * 2);
  bf16_t* W1b    = (bf16_t*)carve((size_t)PRE * KP * 2);
  bf16_t* W2b    = (bf16_t*)carve((size_t)PRE * PRE * 2);
  bf16_t* Wpb    = (bf16_t*)carve((size_t)NMEL * KOUT * 2);
  bf16_t* decp   = (bf16_t*)carve((size_t)TT * BB * KP * 2);
  bf16_t* a1     = (bf16_t*)carve((size_t)TT * BB * PRE * 2);
  bf16_t* xcat   = (bf16_t*)carve((size_t)TT * BB * XC * 2);
  bf16_t* h2hist = (bf16_t*)carve((size_t)TT * BB * HH * 2);
  bf16_t* h1b    = (bf16_t*)carve((size_t)BB * HH * 2);
  bf16_t* h2b    = (bf16_t*)carve((size_t)BB * HH * 2);
  float*  c1     = (float*)carve((size_t)BB * HH * 4);
  float*  c2     = (float*)carve((size_t)BB * HH * 4);
  float*  gates  = (float*)carve((size_t)BB * G4 * 4);

  dim3 blk(BT);
  // weight + input conversion to bf16 (L2-resident reuse across 800 steps)
  k_cvt<<<512, blk, 0, stream>>>(Wih1, Wih1b, G4 * XC);
  k_cvt<<<512, blk, 0, stream>>>(Whh1, Whh1b, G4 * HH);
  k_cvt<<<512, blk, 0, stream>>>(Wih2, Wih2b, G4 * HH);
  k_cvt<<<512, blk, 0, stream>>>(Whh2, Whh2b, G4 * HH);
  k_cvt<<<64, blk, 0, stream>>>(W2, W2b, PRE * PRE);
  k_cvt<<<64, blk, 0, stream>>>(Wp, Wpb, NMEL * KOUT);
  k_cvt_w1<<<64, blk, 0, stream>>>(W1, W1b);
  k_prep_dec<<<512, blk, 0, stream>>>(p_dec, decp);
  k_prep_dur<<<1024, blk, 0, stream>>>(p_dur, xcat);

  // parallel prenet (WMMA)
  k_prenet1<<<256, blk, 0, stream>>>(decp, W1b, pm1, a1);
  k_prenet2<<<256, blk, 0, stream>>>(a1, W2b, pm2, xcat);

  // persistent sequential decoder (WMMA + grid-wide barriers)
  k_decoder<<<NB, blk, 0, stream>>>(xcat, Wih1b, Whh1b, Wih2b, Whh2b, b1, g1,
                                    be1, b2, g2, be2, dm1, dm2, h1b, h2b, c1,
                                    c2, gates, h2hist);

  // parallel output projection (WMMA)
  k_outproj<<<256, blk, 0, stream>>>(h2hist, xcat, Wpb, bp, out);
}
